// DAG_61246233641129
// MI455X (gfx1250) — matrix operand order repaired and checked by reading, not compiled, not run on hardware
//
#include <hip/hip_runtime.h>

typedef __attribute__((ext_vector_type(16))) _Float16 v16h;
typedef __attribute__((ext_vector_type(8)))  _Float16 v8h;
typedef __attribute__((ext_vector_type(2)))  _Float16 h2;
typedef __attribute__((ext_vector_type(8)))  float    v8f;

#define IN_SIZE 1024
#define N_NODES 4224
#define TOTALW  5248
#define BATCH   512
#define LDACC   4224

// ---------------------------------------------------------------------------
// f32 -> f16 convert (for the input activations)
// ---------------------------------------------------------------------------
__global__ void cvt_f32_f16_kernel(const float* __restrict__ src,
                                   _Float16* __restrict__ dst, int n) {
  int i = blockIdx.x * 256 + threadIdx.x;
  if (i < n) dst[i] = (_Float16)src[i];
}

// ---------------------------------------------------------------------------
// WMMA GEMM: acc[M0.., nodeBase..] (+)= A_f16[512 x K] * W[node][kcol..kcol+K)^T
//   grid.x : batch tiles of 128 (4 total)
//   grid.y : node tiles of 128 starting at nodeStart
//   block  : 256 threads = 8 wave32, each wave owns 32(M) x 64(N):
//            2 x 4 fragments of v_wmma_f32_16x16x32_f16
// ---------------------------------------------------------------------------
__global__ __launch_bounds__(256) void gemm_wmma_kernel(
    const _Float16* __restrict__ A, int lda, int K,
    const float* __restrict__ W, int kcol, int nodeStart,
    float* __restrict__ acc, int accumulate)
{
  __shared__ __align__(16) _Float16 As[128 * 32];  // As[m][k]
  __shared__ __align__(16) _Float16 Bs[128 * 32];  // Bs[n][k]

  const int tid   = threadIdx.x;
  const int lane  = tid & 31;
  const int wave  = tid >> 5;
  const int waveM = wave & 3;   // M offset = 32*waveM
  const int waveN = wave >> 2;  // N offset = 64*waveN
  const int M0       = blockIdx.x * 128;
  const int nodeBase = nodeStart + blockIdx.y * 128;

  const int cn = lane & 15;   // column-in-tile for B/C/D, row-in-tile for A
  const int cg = lane >> 4;   // lane group

  v8f C[2][4];
  if (accumulate) {
    #pragma unroll
    for (int mi = 0; mi < 2; ++mi)
      #pragma unroll
      for (int ni = 0; ni < 4; ++ni) {
        const int mb = M0 + waveM * 32 + mi * 16 + 8 * cg;      // C row = r + 8g
        const int nb = nodeBase + waveN * 64 + ni * 16 + cn;    // C col = lane&15
        #pragma unroll
        for (int r = 0; r < 8; ++r)
          C[mi][ni][r] = acc[(size_t)(mb + r) * LDACC + nb];
      }
  } else {
    #pragma unroll
    for (int mi = 0; mi < 2; ++mi)
      #pragma unroll
      for (int ni = 0; ni < 4; ++ni)
        #pragma unroll
        for (int r = 0; r < 8; ++r) C[mi][ni][r] = 0.0f;
  }

  const int ldRow = tid >> 1;        // 0..127
  const int ldSeg = (tid & 1) * 16;  // 0 or 16 (k segment)

  for (int k0 = 0; k0 < K; k0 += 32) {
    // ---- stage A tile (already f16, contiguous along k) ----
    {
      const v8h* gp = (const v8h*)(A + (size_t)(M0 + ldRow) * lda + k0 + ldSeg);
      *(v8h*)&As[ldRow * 32 + ldSeg]     = gp[0];
      *(v8h*)&As[ldRow * 32 + ldSeg + 8] = gp[1];
    }
    // ---- stage B tile from W (f32 -> f16 on the fly) ----
    {
      const float* wp = W + (size_t)(nodeBase + ldRow) * TOTALW + kcol + k0 + ldSeg;
      #pragma unroll
      for (int j = 0; j < 4; ++j) {
        float4 f = ((const float4*)wp)[j];
        _Float16* dst = &Bs[ldRow * 32 + ldSeg + j * 4];
        dst[0] = (_Float16)f.x; dst[1] = (_Float16)f.y;
        dst[2] = (_Float16)f.z; dst[3] = (_Float16)f.w;
      }
      if (k0 + 32 < K) __builtin_prefetch(wp + 32, 0, 0);  // global_prefetch_b8
    }
    __syncthreads();

    // ---- build fragments in contiguous k order (k = 16*cg + s) ----
    v16h a[2], b[4];
    #pragma unroll
    for (int mi = 0; mi < 2; ++mi) {
      const int mrow = waveM * 32 + mi * 16 + cn;  // A: lane m = lane&15
      #pragma unroll
      for (int r = 0; r < 8; ++r) {
        const int kb = 2 * r + 16 * cg;
        h2 p = *(const h2*)&As[mrow * 32 + kb];
        a[mi][2 * r]     = p[0];
        a[mi][2 * r + 1] = p[1];
      }
    }
    #pragma unroll
    for (int ni = 0; ni < 4; ++ni) {
      const int nrow = waveN * 64 + ni * 16 + cn;  // B: lane holds one column
      const v8h* bp = (const v8h*)&Bs[nrow * 32 + 16 * cg];
      v8h lo = bp[0], hi = bp[1];
      #pragma unroll
      for (int h = 0; h < 8; ++h) { b[ni][h] = lo[h]; b[ni][8 + h] = hi[h]; }
    }

    #pragma unroll
    for (int mi = 0; mi < 2; ++mi)
      #pragma unroll
      for (int ni = 0; ni < 4; ++ni)
        C[mi][ni] = __builtin_amdgcn_wmma_f32_16x16x32_f16(
            false, a[mi], false, b[ni], (short)0, C[mi][ni], false, false);

    __syncthreads();
  }

  // ---- store D ----
  #pragma unroll
  for (int mi = 0; mi < 2; ++mi)
    #pragma unroll
    for (int ni = 0; ni < 4; ++ni) {
      const int mb = M0 + waveM * 32 + mi * 16 + 8 * cg;
      const int nb = nodeBase + waveN * 64 + ni * 16 + cn;
      #pragma unroll
      for (int r = 0; r < 8; ++r)
        acc[(size_t)(mb + r) * LDACC + nb] = C[mi][ni][r];
    }
}

// ---------------------------------------------------------------------------
// In-panel triangular recurrence: resolves the 128 serial node dependencies
// of panel `blk`. One thread per batch row; panel weights + running y in LDS.
// Last panel (blk==32) emits sigmoid(tanh(.)) directly to d_out.
// ---------------------------------------------------------------------------
__global__ __launch_bounds__(128) void tri_kernel(
    const float* __restrict__ W, float* __restrict__ acc,
    _Float16* __restrict__ Yh, float* __restrict__ out, int blk)
{
  __shared__ _Float16 wt[128 * 128];   // 32 KB  wt[i][t] = W[jb+i][1024+jb+t]
  __shared__ _Float16 ysh[128 * 128];  // 32 KB  ysh[t][tid] (thread-private column)

  const int tid = threadIdx.x;
  const int r   = blockIdx.x * 128 + tid;   // batch row
  const int jb  = blk * 128;                // first node of panel
  const int colBase = IN_SIZE + jb;

  for (int idx = tid; idx < 128 * 128; idx += 128) {
    const int i = idx >> 7, t = idx & 127;
    wt[idx] = (_Float16)W[(size_t)(jb + i) * TOTALW + colBase + t];
  }
  __syncthreads();

  const float* accRow = acc + (size_t)r * LDACC + jb;
  const bool last = (blk == 32);

  for (int i = 0; i < 128; ++i) {
    float v = accRow[i];
    const _Float16* wrow = &wt[i * 128];
    for (int t = 0; t < i; ++t)            // strictly t < i: future slots are zero
      v += (float)ysh[t * 128 + tid] * (float)wrow[t];
    v = tanhf(v);
    ysh[i * 128 + tid] = (_Float16)v;
    if (last) out[(size_t)r * 128 + i] = 1.0f / (1.0f + expf(-v));
    else      Yh[(size_t)r * 128 + i] = (_Float16)v;
  }
}

// ---------------------------------------------------------------------------
// Host: fixed, deterministic, graph-capturable launch sequence.
// ---------------------------------------------------------------------------
extern "C" void kernel_launch(void* const* d_in, const int* in_sizes, int n_in,
                              void* d_out, int out_size, void* d_ws, size_t ws_size,
                              hipStream_t stream) {
  const float* x = (const float*)d_in[0];   // [512,1024] f32
  const float* W = (const float*)d_in[1];   // [4224,5248] f32
  float* out = (float*)d_out;               // [512,128] f32
  (void)in_sizes; (void)n_in; (void)out_size; (void)ws_size;

  char* ws = (char*)d_ws;
  float*    acc = (float*)ws;                                   // 512*4224*4  = 8,650,752 B
  _Float16* xh  = (_Float16*)(ws + 8650752);                    // 512*1024*2  = 1,048,576 B
  _Float16* Yh  = (_Float16*)(ws + 8650752 + 1048576);          // 512*128*2   =   131,072 B

  // 1) x -> f16
  cvt_f32_f16_kernel<<<(BATCH * IN_SIZE) / 256, 256, 0, stream>>>(x, xh, BATCH * IN_SIZE);

  // 2) acc = x @ W[:, :1024]^T   (K=1024, all 33 node tiles, overwrite)
  gemm_wmma_kernel<<<dim3(4, 33), 256, 0, stream>>>(xh, IN_SIZE, IN_SIZE, W,
                                                    /*kcol=*/0, /*nodeStart=*/0,
                                                    acc, /*accumulate=*/0);

  // 3) 33 panels: serial triangular resolve + rank-128 WMMA forward update
  for (int b = 0; b < 33; ++b) {
    tri_kernel<<<4, 128, 0, stream>>>(W, acc, Yh, out, b);
    if (b < 32) {
      const int nodeStart = 128 * (b + 1);
      gemm_wmma_kernel<<<dim3(4, (N_NODES - nodeStart) / 128), 256, 0, stream>>>(
          Yh, 128, 128, W, /*kcol=*/IN_SIZE + 128 * b, nodeStart, acc, /*accumulate=*/1);
    }
  }
}
